// SAModule_56178172232032
// MI455X (gfx1250) — matrix operand order repaired and checked by reading, not compiled, not run on hardware
//
#include <hip/hip_runtime.h>
#include <hip/hip_bf16.h>

typedef _Float16 v16h __attribute__((ext_vector_type(16)));
typedef _Float16 v8h  __attribute__((ext_vector_type(8)));
typedef float    v8f  __attribute__((ext_vector_type(8)));

#define NB 8
#define NPTS 8192
#define NPOINT 2048
#define NSAMPLE 32
#define POSB 65536            // NSAMPLE*NPOINT positions per batch
#define FLTMAX 3.402823466e+38f

// ---------------- workspace layout (all 256B aligned) ----------------
static const size_t OFF_KNN   = 0;                          // 8*2048*32 int = 2 MB
static const size_t OFF_W0A   = 2097152;                    // 16x96  f16
static const size_t OFF_W0B   = OFF_W0A + 3072;             // 64x32  f16
static const size_t OFF_W1A   = OFF_W0B + 4096;             // 16x64  f16
static const size_t OFF_W1B   = OFF_W1A + 2048;             // 64x32  f16
static const size_t OFF_W2A   = OFF_W1B + 4096;             // 32x64  f16
static const size_t OFF_W2B   = OFF_W2A + 4096;             // 128x32 f16
static const size_t OFF_STATS = OFF_W2B + 8192;             // 64 pairs f32
static const size_t OFF_BUFA  = OFF_STATS + 512;            // [B][POSB][96]  f16
static const size_t OFF_BUFT  = OFF_BUFA + (size_t)NB*POSB*96*2;   // [..][32] f16
static const size_t OFF_BUFB  = OFF_BUFT + (size_t)NB*POSB*32*2;   // [..][128] f16

// ---------------- FPS: one block per batch, iterative argmax ----------------
__global__ __launch_bounds__(1024)
void fps_kernel(const float* __restrict__ xyz, float* __restrict__ new_xyz)
{
    __shared__ float sd[NPTS];
    __shared__ float rv[1024];
    __shared__ int   ri[1024];
    const int b = blockIdx.x;
    const int t = threadIdx.x;
    const float* X = xyz + (size_t)b * NPTS * 3;
    for (int i = t; i < NPTS; i += 1024) sd[i] = FLTMAX;
    if (t == 0) {
        new_xyz[((size_t)b*NPOINT)*3+0] = X[0];
        new_xyz[((size_t)b*NPOINT)*3+1] = X[1];
        new_xyz[((size_t)b*NPOINT)*3+2] = X[2];
    }
    __syncthreads();
    int last = 0;
    for (int k = 1; k < NPOINT; ++k) {
        float lx = X[last*3+0], ly = X[last*3+1], lz = X[last*3+2];
        float bv = -1.0f; int bi = 0;
        for (int i = t; i < NPTS; i += 1024) {
            float dx = X[i*3+0]-lx, dy = X[i*3+1]-ly, dz = X[i*3+2]-lz;
            float d = dx*dx + dy*dy + dz*dz;
            float m = fminf(sd[i], d);
            sd[i] = m;
            if (m > bv) { bv = m; bi = i; }
        }
        rv[t] = bv; ri[t] = bi;
        __syncthreads();
        for (int s = 512; s > 0; s >>= 1) {
            if (t < s) {
                float v2 = rv[t+s]; int i2 = ri[t+s];
                if (v2 > rv[t] || (v2 == rv[t] && i2 < ri[t])) { rv[t] = v2; ri[t] = i2; }
            }
            __syncthreads();
        }
        last = ri[0];
        __syncthreads();
        if (t == 0) {
            new_xyz[((size_t)b*NPOINT+k)*3+0] = X[last*3+0];
            new_xyz[((size_t)b*NPOINT+k)*3+1] = X[last*3+1];
            new_xyz[((size_t)b*NPOINT+k)*3+2] = X[last*3+2];
        }
    }
}

// ---------------- kNN: one block per query, 32 argmin rounds in LDS ----------------
__global__ __launch_bounds__(256)
void knn_kernel(const float* __restrict__ xyz, const float* __restrict__ new_xyz,
                int* __restrict__ knn)
{
    __shared__ float d2[NPTS];
    __shared__ float rv[256];
    __shared__ int   ri[256];
    const int b = blockIdx.x >> 11;
    const int p = blockIdx.x & 2047;
    const int t = threadIdx.x;
    const float* q = new_xyz + ((size_t)b*NPOINT + p)*3;
    const float qx = q[0], qy = q[1], qz = q[2];
    const float* X = xyz + (size_t)b * NPTS * 3;
    for (int i = t; i < NPTS; i += 256) {
        float dx = X[i*3+0]-qx, dy = X[i*3+1]-qy, dz = X[i*3+2]-qz;
        d2[i] = dx*dx + dy*dy + dz*dz;
    }
    __syncthreads();
    int* out = knn + (((size_t)b*NPOINT + p) << 5);
    for (int j = 0; j < NSAMPLE; ++j) {
        float bv = FLTMAX; int bi = 0;
        for (int i = t; i < NPTS; i += 256) {
            float v = d2[i];
            if (v < bv) { bv = v; bi = i; }
        }
        rv[t] = bv; ri[t] = bi;
        __syncthreads();
        for (int s = 128; s > 0; s >>= 1) {
            if (t < s) {
                float v2 = rv[t+s]; int i2 = ri[t+s];
                if (v2 < rv[t] || (v2 == rv[t] && i2 < ri[t])) { rv[t] = v2; ri[t] = i2; }
            }
            __syncthreads();
        }
        if (t == 0) { out[j] = ri[0]; d2[ri[0]] = FLTMAX; }
        __syncthreads();
    }
}

// ---------------- grouping: build f16 activation [b][pos][96] ----------------
__global__ __launch_bounds__(256)
void group_kernel(const float* __restrict__ xyz, const float* __restrict__ feat,
                  const float* __restrict__ new_xyz, const int* __restrict__ knn,
                  _Float16* __restrict__ bufA)
{
    size_t idx = (size_t)blockIdx.x * 256 + threadIdx.x;   // B*NPOINT*NSAMPLE
    int s = idx & 31;
    int k = (idx >> 5) & 2047;
    int b = (int)(idx >> 16);
    int pi = knn[(((size_t)b*NPOINT + k) << 5) + s];
    const float* nx = new_xyz + ((size_t)b*NPOINT + k)*3;
    const float* px = xyz + ((size_t)b*NPTS + pi)*3;
    const float* f  = feat + ((size_t)b*NPTS + pi)*64;
    _Float16* row = bufA + (((size_t)b << 16) + (size_t)s*NPOINT + k) * 96;
    row[0] = (_Float16)(px[0]-nx[0]);
    row[1] = (_Float16)(px[1]-nx[1]);
    row[2] = (_Float16)(px[2]-nx[2]);
    #pragma unroll 8
    for (int j = 0; j < 64; ++j) row[3+j] = (_Float16)f[j];
    #pragma unroll
    for (int j = 67; j < 96; ++j) row[j] = (_Float16)0.0f;
}

// ---------------- weight pad/convert to f16 fragments ----------------
__global__ void padw_kernel(const float* __restrict__ w, _Float16* __restrict__ o,
                            int Cout, int Cin, int Cinp)
{
    int i = blockIdx.x * 256 + threadIdx.x;
    int tot = Cout * Cinp;
    if (i >= tot) return;
    int m = i / Cinp, k = i % Cinp;
    o[i] = (k < Cin) ? (_Float16)w[m*Cin + k] : (_Float16)0.0f;
}

__global__ void zero_stats_kernel(float* stats)
{
    if (threadIdx.x < 128) stats[threadIdx.x] = 0.0f;
}

// ---------------- WMMA GEMM: out[pos][Cout] = W[Cout][Cin] * act[pos][Cin]^T ----------------
// act layout [b][pos][CINP] halves; B-fragment = contiguous 32B per lane;
// D store = 8 contiguous channels per lane. Optional fused group-norm statistics.
template<int CINP, int COUT, int COUTP, int G, bool STATS>
__global__ __launch_bounds__(128)
void gemm_kernel(const _Float16* __restrict__ act, const _Float16* __restrict__ Wp,
                 const float* __restrict__ bias, _Float16* __restrict__ out,
                 float* __restrict__ stats)
{
    constexpr int KT = CINP / 32;
    constexpr int MT = COUT / 16;
    static_assert(!STATS || (COUT / G == 16), "group size must be 16");
    const int lane = threadIdx.x & 31;
    const int wave = threadIdx.x >> 5;
    const int l2   = lane & 15;
    const int hi   = lane >> 4;
    const long long tile = (long long)blockIdx.x * 4 + wave;   // B*4096 tiles total
    const int bidx = (int)(tile >> 12);
    const int nt   = (int)(tile & 4095);
    const long long row = ((long long)bidx << 16) + (long long)nt*16 + l2;
    const _Float16* arow = act + row * CINP;

    // B fragments: K x N (32x16) per K-tile; lane<16 holds K 0..15 of column l2,
    // lane>=16 holds K 16..31 (this code's order, same as A).
    v16h bfrag[KT];
    #pragma unroll
    for (int kt = 0; kt < KT; ++kt)
        bfrag[kt] = *(const v16h*)(arow + kt*32 + hi*16);

    __builtin_prefetch(Wp, 0, 3);

    #pragma unroll
    for (int mt = 0; mt < MT; ++mt) {
        v8f acc;
        #pragma unroll
        for (int r = 0; r < 8; ++r) acc[r] = bias[mt*16 + hi*8 + r];
        #pragma unroll
        for (int kt = 0; kt < KT; ++kt) {
            // A fragment: M=l2 row of weights; lane-lo K {0..15}, lane-hi K {16..31}
            const _Float16* wrow = Wp + (size_t)(mt*16 + l2)*CINP + kt*32;
            v8h a0 = *(const v8h*)(wrow + hi*16);
            v8h a1 = *(const v8h*)(wrow + hi*16 + 8);
            v16h afrag = __builtin_shufflevector(a0, a1,
                0,1,2,3,4,5,6,7,8,9,10,11,12,13,14,15);
            acc = __builtin_amdgcn_wmma_f32_16x16x32_f16(
                false, afrag, false, bfrag[kt], (short)0, acc, false, false);
        }
        // D: lane covers channels mt*16 + hi*8 .. +7 at column l2 -> contiguous f16 store
        _Float16* orow = out + row * COUTP + mt*16 + hi*8;
        v8h hs;
        #pragma unroll
        for (int r = 0; r < 8; ++r) hs[r] = (_Float16)acc[r];
        *(v8h*)orow = hs;

        if constexpr (STATS) {
            float s = 0.f, q = 0.f;
            #pragma unroll
            for (int r = 0; r < 8; ++r) { float v = acc[r]; s += v; q += v*v; }
            #pragma unroll
            for (int off = 16; off > 0; off >>= 1) {
                s += __shfl_xor(s, off, 32);
                q += __shfl_xor(q, off, 32);
            }
            if (lane == 0) {
                atomicAdd(&stats[(bidx*G + mt)*2 + 0], s);
                atomicAdd(&stats[(bidx*G + mt)*2 + 1], q);
            }
        }
    }
    if constexpr (COUTP > COUT) {
        v8h z = {};
        for (int c = COUT + hi*8; c < COUTP; c += 16)
            *(v8h*)(out + row * COUTP + c) = z;
    }
}

__global__ void finalize_kernel(float* stats, int n, float invcnt)
{
    int i = threadIdx.x;
    if (i < n) {
        float s = stats[2*i], q = stats[2*i+1];
        float m = s * invcnt;
        float v = q * invcnt - m*m;
        stats[2*i]   = m;
        stats[2*i+1] = rsqrtf(v + 1e-5f);
    }
}

// ---------------- group-norm apply + ReLU (in place, f16) ----------------
template<int C, int G>
__global__ __launch_bounds__(256)
void norm_kernel(_Float16* __restrict__ buf, const float* __restrict__ stats,
                 const float* __restrict__ gamma, const float* __restrict__ beta)
{
    constexpr int CH = C / 8;
    long long idx = (long long)blockIdx.x * 256 + threadIdx.x;
    long long total = (long long)NB * POSB * CH;
    if (idx >= total) return;
    int chunk = (int)(idx % CH);
    long long r = idx / CH;
    int b = (int)(r >> 16);
    int g = (chunk * 8) / 16;
    float mean = stats[(b*G + g)*2 + 0];
    float rstd = stats[(b*G + g)*2 + 1];
    _Float16* p = buf + r * C + chunk*8;
    v8h h = *(v8h*)p;
    #pragma unroll
    for (int j = 0; j < 8; ++j) {
        int c = chunk*8 + j;
        float f = (float)h[j];
        f = (f - mean) * rstd * gamma[c] + beta[c];
        h[j] = (_Float16)fmaxf(f, 0.0f);
    }
    *(v8h*)p = h;
}

// ---------------- max over samples -> new_features (f32) ----------------
__global__ __launch_bounds__(256)
void maxpool_kernel(const _Float16* __restrict__ buf, float* __restrict__ outF)
{
    long long idx = (long long)blockIdx.x * 256 + threadIdx.x;  // B*NPOINT*16
    int chunk = (int)(idx & 15);
    int p = (int)((idx >> 4) & 2047);
    int b = (int)(idx >> 15);
    float m[8];
    #pragma unroll
    for (int r = 0; r < 8; ++r) m[r] = -FLTMAX;
    for (int s = 0; s < NSAMPLE; ++s) {
        v8h h = *(const v8h*)(buf + (((long long)b << 16) + (long long)s*NPOINT + p)*128 + chunk*8);
        #pragma unroll
        for (int r = 0; r < 8; ++r) m[r] = fmaxf(m[r], (float)h[r]);
    }
    float* o = outF + (size_t)NB*NPOINT*3 + (((long long)b*NPOINT + p)*128) + chunk*8;
    #pragma unroll
    for (int r = 0; r < 8; ++r) o[r] = m[r];
}

// ---------------- launch ----------------
extern "C" void kernel_launch(void* const* d_in, const int* in_sizes, int n_in,
                              void* d_out, int out_size, void* d_ws, size_t ws_size,
                              hipStream_t stream)
{
    const float* xyz  = (const float*)d_in[0];
    const float* feat = (const float*)d_in[1];
    const float* w0a = (const float*)d_in[2],  *b0a = (const float*)d_in[3];
    const float* w0b = (const float*)d_in[4],  *b0b = (const float*)d_in[5];
    const float* g0  = (const float*)d_in[6],  *be0 = (const float*)d_in[7];
    const float* w1a = (const float*)d_in[8],  *b1a = (const float*)d_in[9];
    const float* w1b = (const float*)d_in[10], *b1b = (const float*)d_in[11];
    const float* g1  = (const float*)d_in[12], *be1 = (const float*)d_in[13];
    const float* w2a = (const float*)d_in[14], *b2a = (const float*)d_in[15];
    const float* w2b = (const float*)d_in[16], *b2b = (const float*)d_in[17];
    const float* g2  = (const float*)d_in[18], *be2 = (const float*)d_in[19];

    float* outF = (float*)d_out;
    char* W = (char*)d_ws;
    int*       knnIdx = (int*)(W + OFF_KNN);
    _Float16*  W0A = (_Float16*)(W + OFF_W0A);
    _Float16*  W0B = (_Float16*)(W + OFF_W0B);
    _Float16*  W1A = (_Float16*)(W + OFF_W1A);
    _Float16*  W1B = (_Float16*)(W + OFF_W1B);
    _Float16*  W2A = (_Float16*)(W + OFF_W2A);
    _Float16*  W2B = (_Float16*)(W + OFF_W2B);
    float*     stats = (float*)(W + OFF_STATS);
    _Float16*  bufA = (_Float16*)(W + OFF_BUFA);
    _Float16*  bufT = (_Float16*)(W + OFF_BUFT);
    _Float16*  bufB = (_Float16*)(W + OFF_BUFB);

    const int GEMM_BLOCKS = (NB * (POSB/16)) / 4;   // 4 waves/block, 1 tile/wave
    const float invcnt = 1.0f / (16.0f * (float)POSB);

    // weights -> padded f16
    padw_kernel<<<6,  256, 0, stream>>>(w0a, W0A, 16, 67, 96);
    padw_kernel<<<8,  256, 0, stream>>>(w0b, W0B, 64, 16, 32);
    padw_kernel<<<4,  256, 0, stream>>>(w1a, W1A, 16, 64, 64);
    padw_kernel<<<8,  256, 0, stream>>>(w1b, W1B, 64, 16, 32);
    padw_kernel<<<8,  256, 0, stream>>>(w2a, W2A, 32, 64, 64);
    padw_kernel<<<16, 256, 0, stream>>>(w2b, W2B, 128, 32, 32);

    // sampling / grouping
    fps_kernel  <<<NB, 1024, 0, stream>>>(xyz, outF);
    knn_kernel  <<<NB*NPOINT, 256, 0, stream>>>(xyz, outF, knnIdx);
    group_kernel<<<(NB*NPOINT*NSAMPLE)/256, 256, 0, stream>>>(xyz, feat, outF, knnIdx, bufA);

    // layer 0: 96 -> 16 (pad 32) -> 64, GN(G=4)+ReLU
    gemm_kernel<96, 16, 32, 1, false><<<GEMM_BLOCKS, 128, 0, stream>>>(bufA, W0A, b0a, bufT, nullptr);
    zero_stats_kernel<<<1, 256, 0, stream>>>(stats);
    gemm_kernel<32, 64, 64, 4, true><<<GEMM_BLOCKS, 128, 0, stream>>>(bufT, W0B, b0b, bufB, stats);
    finalize_kernel<<<1, 64, 0, stream>>>(stats, NB*4, invcnt);
    norm_kernel<64, 4><<<(NB*POSB*8)/256, 256, 0, stream>>>(bufB, stats, g0, be0);

    // layer 1: 64 -> 16 (pad 32) -> 64, GN(G=4)+ReLU
    gemm_kernel<64, 16, 32, 1, false><<<GEMM_BLOCKS, 128, 0, stream>>>(bufB, W1A, b1a, bufT, nullptr);
    zero_stats_kernel<<<1, 256, 0, stream>>>(stats);
    gemm_kernel<32, 64, 64, 4, true><<<GEMM_BLOCKS, 128, 0, stream>>>(bufT, W1B, b1b, bufA, stats);
    finalize_kernel<<<1, 64, 0, stream>>>(stats, NB*4, invcnt);
    norm_kernel<64, 4><<<(NB*POSB*8)/256, 256, 0, stream>>>(bufA, stats, g1, be1);

    // layer 2: 64 -> 32 -> 128, GN(G=8)+ReLU
    gemm_kernel<64, 32, 32, 1, false><<<GEMM_BLOCKS, 128, 0, stream>>>(bufA, W2A, b2a, bufT, nullptr);
    zero_stats_kernel<<<1, 256, 0, stream>>>(stats);
    gemm_kernel<32, 128, 128, 8, true><<<GEMM_BLOCKS, 128, 0, stream>>>(bufT, W2B, b2b, bufB, stats);
    finalize_kernel<<<1, 64, 0, stream>>>(stats, NB*8, invcnt);
    norm_kernel<128, 8><<<(NB*POSB*16)/256, 256, 0, stream>>>(bufB, stats, g2, be2);

    // max over samples -> (B, NPOINT, 128) f32
    maxpool_kernel<<<(NB*NPOINT*16)/256, 256, 0, stream>>>(bufB, outF);
}
